// SecondEncoder_16939351016215
// MI455X (gfx1250) — compile-verified, matrix operand order checked by reading, not run on hardware
//
#include <hip/hip_runtime.h>
#include <math.h>
#include <stdint.h>

#ifndef __has_builtin
#define __has_builtin(x) 0
#endif
#if __has_builtin(__builtin_amdgcn_tensor_load_to_lds)
#define HAVE_TDM 1
#else
#define HAVE_TDM 0
#endif

// ---------------------------------------------------------------------------
// Types for CDNA5 WMMA (wave32): v_wmma_f32_16x16x32_bf16
// ---------------------------------------------------------------------------
typedef __attribute__((ext_vector_type(16))) __bf16 v16bf;
typedef __attribute__((ext_vector_type(8)))  float  v8f;
typedef __attribute__((ext_vector_type(4)))  unsigned int v4u;
typedef __attribute__((ext_vector_type(8)))  int v8i;
typedef __attribute__((ext_vector_type(4)))  int v4i;

union FragU { v16bf v; v4u q[2]; };

// A/B fragment for 16-bit 16x32 (or N x K): per lane two contiguous 16B loads.
__device__ __forceinline__ v16bf load_frag(const __bf16* __restrict__ base,
                                           int ld, int row, int k0, int lane) {
  const int r  = row + (lane & 15);
  const int kk = k0 + ((lane >> 4) << 3);
  FragU u;
  u.q[0] = *reinterpret_cast<const v4u*>(base + (size_t)r * ld + kk);
  u.q[1] = *reinterpret_cast<const v4u*>(base + (size_t)r * ld + kk + 16);
  return u.v;
}

// Same fragment, but fp32 source (LDS softmax probabilities) -> convert.
__device__ __forceinline__ v16bf frag_from_f32(const float* __restrict__ s,
                                               int ld, int row, int k0, int lane) {
  const int r  = row + (lane & 15);
  const int kk = k0 + ((lane >> 4) << 3);
  const float* p = s + (size_t)r * ld + kk;
  v16bf v;
#pragma unroll
  for (int i = 0; i < 8; ++i) v[i]     = (__bf16)p[i];
#pragma unroll
  for (int i = 0; i < 8; ++i) v[8 + i] = (__bf16)p[16 + i];
  return v;
}

__device__ __forceinline__ v8f wmma_bf16(v16bf a, v16bf b, v8f c) {
  return __builtin_amdgcn_wmma_f32_16x16x32_bf16(false, a, false, b,
                                                 (short)0, c, false, false);
}

// ---------------------------------------------------------------------------
// Tensor Data Mover: 2-D bf16 tile (rows of 512 elems) global -> LDS with
// row padding of 4 DWORDs (pad_interval=7: every 256 DWORDs) so LDS row
// stride is 520 elems -> conflict-free ds_load_b128 fragment reads.
// ---------------------------------------------------------------------------
__device__ __forceinline__ void wait_tensorcnt0() {
#if __has_builtin(__builtin_amdgcn_s_wait_tensorcnt)
  __builtin_amdgcn_s_wait_tensorcnt(0);
#else
  asm volatile("s_wait_tensorcnt 0x0" ::: "memory");
#endif
}

#if HAVE_TDM
__device__ __forceinline__ void tdm_load_rows512_bf16(unsigned lds_addr,
                                                      const void* gaddr,
                                                      unsigned rows) {
  const unsigned long long ga = (unsigned long long)(uintptr_t)gaddr;
  v4u g0 = {0u, 0u, 0u, 0u};
  g0[0] = 1u;                                   // count=1, user descriptor
  g0[1] = lds_addr;                             // LDS byte address
  g0[2] = (unsigned)ga;                         // global_addr[31:0]
  g0[3] = (unsigned)((ga >> 32) & 0x1FFFFFFu) | (2u << 30);  // addr[56:32]|type=2
  v8i g1 = {0, 0, 0, 0, 0, 0, 0, 0};
  // data_size=2B, pad_enable, pad_interval=7 (256 dw), pad_amount=3 (4 dw)
  g1[0] = (int)((1u << 16) | (1u << 20) | (7u << 22) | (3u << 25));
  g1[1] = (int)(512u << 16);                    // tensor_dim0 lo16 = 512
  g1[2] = (int)((rows & 0xFFFFu) << 16);        // tensor_dim0 hi=0 | tensor_dim1 lo
  g1[3] = (int)(512u << 16);                    // tensor_dim1 hi=0 | tile_dim0 = 512
  g1[4] = (int)(rows & 0xFFFFu);                // tile_dim1 = rows, tile_dim2 = 0
  g1[5] = 512;                                  // tensor_dim0_stride = 512 elems
  v4i z4 = {0, 0, 0, 0};
#if __clang_major__ >= 23
  v8i z8 = {0, 0, 0, 0, 0, 0, 0, 0};
  __builtin_amdgcn_tensor_load_to_lds(g0, g1, z4, z4, z8, 0);
#else
  __builtin_amdgcn_tensor_load_to_lds(g0, g1, z4, z4, 0);
#endif
}
#endif

// Fallback (and reference semantics): cooperative padded copy, 16B chunks.
__device__ __forceinline__ void copy_rows512_padded(__bf16* dst,
                                                    const __bf16* src,
                                                    int rows, int tid) {
  const int total = rows * 64;                  // 64 x 16B chunks per row
  for (int c = tid; c < total; c += 256) {
    const int r = c >> 6, w = c & 63;
    *reinterpret_cast<v4u*>(dst + r * 520 + w * 8) =
        *reinterpret_cast<const v4u*>(src + (size_t)r * 512 + w * 8);
  }
}

// ---------------------------------------------------------------------------
// fp32 -> bf16 convert (grid-stride)
// ---------------------------------------------------------------------------
__global__ __launch_bounds__(256) void k_cvt_bf16(const float* __restrict__ s,
                                                  __bf16* __restrict__ d, size_t n) {
  for (size_t i = (size_t)blockIdx.x * 256u + threadIdx.x; i < n;
       i += (size_t)gridDim.x * 256u)
    d[i] = (__bf16)s[i];
}

// ---------------------------------------------------------------------------
// GEMM: C[M,N] = A[M,K=512](bf16) * W[N,K=512](bf16)^T + bias
// Block tile 128x128, 8 waves -> each wave 32(M) x 64(N) = 2x4 WMMA tiles.
// W panel (128 x 512) staged in LDS via TDM (padded rows); A streamed from
// global as b128 fragment loads. Optional fp32 / bf16 / bf16-transposed out.
// ---------------------------------------------------------------------------
__global__ __launch_bounds__(256) void k_gemm_bf16(
    const __bf16* __restrict__ A, const __bf16* __restrict__ W,
    const float* __restrict__ bias,
    float* __restrict__ outF, __bf16* __restrict__ outB, __bf16* __restrict__ outBT,
    int M, int N, int K) {
  __shared__ __bf16 sW[128 * 520];              // 130 KiB (padded rows)
  const int lane = threadIdx.x & 31;
  const int wid  = threadIdx.x >> 5;
  const int bn0 = blockIdx.x * 128;
  const int wm  = blockIdx.y * 128 + (wid >> 1) * 32;
  const int wnl = (wid & 1) * 64;               // N offset inside panel
  const int wn  = bn0 + wnl;

  const __bf16* Wp = W + (size_t)bn0 * K;       // panel: 128 consecutive W rows
#if HAVE_TDM
  if (wid == 0) {
    tdm_load_rows512_bf16((unsigned)(uintptr_t)sW, Wp, 128u);
    wait_tensorcnt0();
  }
#else
  copy_rows512_padded(sW, Wp, 128, threadIdx.x);
#endif
  __syncthreads();

  v8f acc[2][4];
#pragma unroll
  for (int i = 0; i < 2; ++i)
#pragma unroll
    for (int j = 0; j < 4; ++j) acc[i][j] = (v8f)(0.0f);

  for (int k = 0; k < K; k += 32) {
    if (k + 64 < K)  // keep the A stream warm (global_prefetch_b8)
      __builtin_prefetch(A + (size_t)(wm + (lane & 15)) * K + k + 64, 0, 1);
    v16bf a0 = load_frag(A, K, wm,      k, lane);
    v16bf a1 = load_frag(A, K, wm + 16, k, lane);
    v16bf b0 = load_frag(sW, 520, wnl,      k, lane);
    v16bf b1 = load_frag(sW, 520, wnl + 16, k, lane);
    v16bf b2 = load_frag(sW, 520, wnl + 32, k, lane);
    v16bf b3 = load_frag(sW, 520, wnl + 48, k, lane);
    acc[0][0] = wmma_bf16(a0, b0, acc[0][0]);
    acc[0][1] = wmma_bf16(a0, b1, acc[0][1]);
    acc[0][2] = wmma_bf16(a0, b2, acc[0][2]);
    acc[0][3] = wmma_bf16(a0, b3, acc[0][3]);
    acc[1][0] = wmma_bf16(a1, b0, acc[1][0]);
    acc[1][1] = wmma_bf16(a1, b1, acc[1][1]);
    acc[1][2] = wmma_bf16(a1, b2, acc[1][2]);
    acc[1][3] = wmma_bf16(a1, b3, acc[1][3]);
  }

  // C/D layout: lane -> N (col), VGPR i -> M (row), lanes 16-31 rows +8.
  const int rowL = (lane >> 4) << 3;
  const int col  = lane & 15;
#pragma unroll
  for (int mt = 0; mt < 2; ++mt)
#pragma unroll
    for (int nt = 0; nt < 4; ++nt) {
      const int n = wn + nt * 16 + col;
      const float bv = bias ? bias[n] : 0.0f;
#pragma unroll
      for (int i = 0; i < 8; ++i) {
        const int m = wm + mt * 16 + rowL + i;
        const float v = acc[mt][nt][i] + bv;
        if (outF)  outF[(size_t)m * N + n] = v;
        if (outB)  outB[(size_t)m * N + n] = (__bf16)v;
        if (outBT) outBT[(size_t)n * M + m] = (__bf16)v;
      }
    }
}

// ---------------------------------------------------------------------------
// Flash-style attention. One block = (batch, 32 query rows). 8 waves.
// Q tile (32x512 bf16) staged in LDS via TDM with padded rows. Key chunks of
// 128 (wave owns 16 keys in the score GEMM). Softmax stats fp32 in LDS
// (132-float padded rows). P*V: wave owns 64 of 512 output cols; V supplied
// transposed ([D, B*S]) so its B-fragments are K-contiguous global loads.
// ---------------------------------------------------------------------------
__global__ __launch_bounds__(256) void k_attn(
    const __bf16* __restrict__ Q, const __bf16* __restrict__ Km,
    const __bf16* __restrict__ Vt, float* __restrict__ O, int S, int Mtot) {
  constexpr int D  = 512;
  constexpr int CH = 128;
  constexpr int LQ = 520;   // padded bf16 row stride
  constexpr int LS = 132;   // padded fp32 row stride
  __shared__ __bf16 sQ[32 * LQ];
  __shared__ float  sS[32 * LS];
  __shared__ float  sM[32], sL[32], sA[32];

  const int lane = threadIdx.x & 31;
  const int wid  = threadIdx.x >> 5;
  const int b    = blockIdx.y;
  const int q0   = blockIdx.x * 32;

  const __bf16* Qb = Q + ((size_t)b * S + q0) * D;
#if HAVE_TDM
  if (wid == 0) {
    tdm_load_rows512_bf16((unsigned)(uintptr_t)sQ, Qb, 32u);
    wait_tensorcnt0();
  }
#else
  copy_rows512_padded(sQ, Qb, 32, threadIdx.x);
#endif
  if (threadIdx.x < 32) { sM[threadIdx.x] = -3.0e38f; sL[threadIdx.x] = 0.0f; }
  __syncthreads();

  v8f o[2][4];
#pragma unroll
  for (int i = 0; i < 2; ++i)
#pragma unroll
    for (int j = 0; j < 4; ++j) o[i][j] = (v8f)(0.0f);

  const float scale = 0.044194173824f;  // 1/sqrt(512)
  const int rowL = (lane >> 4) << 3;
  const int col  = lane & 15;

  for (int kc = 0; kc < S; kc += CH) {
    // ---- scores: wave computes S[32 x 16] for keys kc + wid*16 .. +15
    v8f sc[2] = {(v8f)(0.0f), (v8f)(0.0f)};
    const __bf16* Kb = Km + ((size_t)b * S + kc + wid * 16) * D;
    for (int k = 0; k < D; k += 32) {
      v16bf q0f = load_frag(sQ, LQ, 0,  k, lane);
      v16bf q1f = load_frag(sQ, LQ, 16, k, lane);
      v16bf kf  = load_frag(Kb, D, 0,  k, lane);
      sc[0] = wmma_bf16(q0f, kf, sc[0]);
      sc[1] = wmma_bf16(q1f, kf, sc[1]);
    }
#pragma unroll
    for (int mt = 0; mt < 2; ++mt)
#pragma unroll
      for (int i = 0; i < 8; ++i)
        sS[(mt * 16 + rowL + i) * LS + wid * 16 + col] = sc[mt][i] * scale;
    __syncthreads();

    // ---- online softmax stats (wave 0, lane = query row)
    if (wid == 0) {
      const float mOld = sM[lane], lOld = sL[lane];
      float mc = -3.0e38f;
      for (int j = 0; j < CH; ++j) mc = fmaxf(mc, sS[lane * LS + j]);
      const float mNew = fmaxf(mOld, mc);
      const float alpha = __expf(mOld - mNew);
      float sum = 0.0f;
      for (int j = 0; j < CH; ++j) {
        const float p = __expf(sS[lane * LS + j] - mNew);
        sS[lane * LS + j] = p;
        sum += p;
      }
      sM[lane] = mNew; sL[lane] = alpha * lOld + sum; sA[lane] = alpha;
    }
    __syncthreads();

    // ---- rescale O, accumulate P * V
#pragma unroll
    for (int mt = 0; mt < 2; ++mt)
#pragma unroll
      for (int i = 0; i < 8; ++i) {
        const float al = sA[mt * 16 + rowL + i];
#pragma unroll
        for (int nt = 0; nt < 4; ++nt) o[mt][nt][i] *= al;
      }
    for (int kk = 0; kk < CH; kk += 32) {
      v16bf p0 = frag_from_f32(sS, LS, 0,  kk, lane);
      v16bf p1 = frag_from_f32(sS, LS, 16, kk, lane);
      const __bf16* Vb = Vt + (size_t)b * S + kc + kk;
#pragma unroll
      for (int nt = 0; nt < 4; ++nt) {
        v16bf vf = load_frag(Vb, Mtot, wid * 64 + nt * 16, 0, lane);
        o[0][nt] = wmma_bf16(p0, vf, o[0][nt]);
        o[1][nt] = wmma_bf16(p1, vf, o[1][nt]);
      }
    }
    __syncthreads();
  }

  // ---- normalize and store fp32 output [B,S,D]
#pragma unroll
  for (int mt = 0; mt < 2; ++mt)
#pragma unroll
    for (int i = 0; i < 8; ++i) {
      const int row = mt * 16 + rowL + i;
      const float invl = 1.0f / sL[row];
#pragma unroll
      for (int nt = 0; nt < 4; ++nt) {
        const int c = wid * 64 + nt * 16 + col;
        O[((size_t)b * S + q0 + row) * D + c] = o[mt][nt][i] * invl;
      }
    }
}

// ---------------------------------------------------------------------------
// LayerNorm (wave per row, D=512 -> 16 elems/lane) with fused residual blend.
// mode 1: y = LN(base + lam*a1 + (1-lam)*a2)   (lam read from device)
// mode 0: y = LN(base + a1)
// ---------------------------------------------------------------------------
__global__ __launch_bounds__(256) void k_ln(
    const float* __restrict__ base, const float* __restrict__ a1,
    const float* __restrict__ a2, const float* __restrict__ lam_ptr, int mode,
    const float* __restrict__ g, const float* __restrict__ bta,
    float* __restrict__ outF, __bf16* __restrict__ outB, int rows) {
  const int row  = blockIdx.x * 8 + (threadIdx.x >> 5);
  const int lane = threadIdx.x & 31;
  if (row >= rows) return;
  float s1 = 1.0f, s2 = 0.0f;
  if (mode == 1) { const float lam = *lam_ptr; s1 = lam; s2 = 1.0f - lam; }

  const size_t off = (size_t)row * 512;
  float x[16];
  float sum = 0.0f;
#pragma unroll
  for (int i = 0; i < 16; ++i) {
    const int c = lane + 32 * i;
    float v = base[off + c];
    if (mode == 1)      v += s1 * a1[off + c] + s2 * a2[off + c];
    else if (a1)        v += a1[off + c];
    x[i] = v; sum += v;
  }
#pragma unroll
  for (int d = 16; d >= 1; d >>= 1) sum += __shfl_xor(sum, d, 32);
  const float mu = sum * (1.0f / 512.0f);
  float vs = 0.0f;
#pragma unroll
  for (int i = 0; i < 16; ++i) { const float dv = x[i] - mu; vs += dv * dv; }
#pragma unroll
  for (int d = 16; d >= 1; d >>= 1) vs += __shfl_xor(vs, d, 32);
  const float inv = rsqrtf(vs * (1.0f / 512.0f) + 1e-5f);
#pragma unroll
  for (int i = 0; i < 16; ++i) {
    const int c = lane + 32 * i;
    const float y = (x[i] - mu) * inv * g[c] + bta[c];
    if (outF) outF[off + c] = y;
    if (outB) outB[off + c] = (__bf16)y;
  }
}

// ---------------------------------------------------------------------------
// Host orchestration
// ---------------------------------------------------------------------------
extern "C" void kernel_launch(void* const* d_in, const int* in_sizes, int n_in,
                              void* d_out, int out_size, void* d_ws, size_t ws_size,
                              hipStream_t stream) {
  (void)in_sizes; (void)n_in; (void)out_size; (void)ws_size;
  constexpr int B = 8, S = 2048, D = 512;
  constexpr int M = B * S;                 // 16384
  const size_t MK  = (size_t)M * D;        // 8,388,608 elems
  const size_t SZB = MK * 2;               // bf16 activation buffer bytes
  const size_t SZF = MK * 4;               // fp32 activation buffer bytes
  const size_t WB  = (size_t)D * D * 2;    // bf16 weight bytes

  const float* x1   = (const float*)d_in[0];
  const float* x2   = (const float*)d_in[1];
  const float* wsrc[8] = {(const float*)d_in[2],  (const float*)d_in[4],
                          (const float*)d_in[6],  (const float*)d_in[8],
                          (const float*)d_in[10], (const float*)d_in[12],
                          (const float*)d_in[19], (const float*)d_in[21]};
  const float* q1_b = (const float*)d_in[3];
  const float* k1_b = (const float*)d_in[5];
  const float* v1_b = (const float*)d_in[7];
  const float* q2_b = (const float*)d_in[9];
  const float* k2_b = (const float*)d_in[11];
  const float* v2_b = (const float*)d_in[13];
  const float* lam  = (const float*)d_in[14];
  const float* ln1g = (const float*)d_in[15];
  const float* ln1b = (const float*)d_in[16];
  const float* ln2g = (const float*)d_in[17];
  const float* ln2b = (const float*)d_in[18];
  const float* fc1b = (const float*)d_in[20];
  const float* fc2b = (const float*)d_in[22];

  char* w = (char*)d_ws;
  size_t off = 0;
  __bf16* wb[8];
  for (int i = 0; i < 8; ++i) { wb[i] = (__bf16*)(w + off); off += WB; }
  __bf16* x1b = (__bf16*)(w + off); off += SZB;
  __bf16* x2b = (__bf16*)(w + off); off += SZB;
  __bf16* Q1  = (__bf16*)(w + off); off += SZB;
  __bf16* K1  = (__bf16*)(w + off); off += SZB;
  __bf16* V1t = (__bf16*)(w + off); off += SZB;
  __bf16* Q2  = (__bf16*)(w + off); off += SZB;
  __bf16* K2  = (__bf16*)(w + off); off += SZB;
  __bf16* V2t = (__bf16*)(w + off); off += SZB;
  float*  xw1 = (float*)(w + off); off += SZF;
  float*  xw2 = (float*)(w + off); off += SZF;
  float*  hf  = (float*)(w + off); off += SZF;
  // aliases (producers run strictly after the aliased buffers are dead):
  __bf16* hb  = Q1;                       // written by ln1 after both attns
  __bf16* h2b = K1;                       // written by ln2
  float*  tb  = (float*)V1t;              // fc1 out (spans V1t+Q2 = SZF bytes)

  // 1) fp32 -> bf16 conversions
  k_cvt_bf16<<<2048, 256, 0, stream>>>(x1, x1b, MK);
  k_cvt_bf16<<<2048, 256, 0, stream>>>(x2, x2b, MK);
  for (int i = 0; i < 8; ++i)
    k_cvt_bf16<<<256, 256, 0, stream>>>(wsrc[i], wb[i], (size_t)D * D);

  // 2) QKV projections (V stored transposed for attention B-fragments)
  dim3 gg(D / 128, M / 128);
  k_gemm_bf16<<<gg, 256, 0, stream>>>(x1b, wb[0], q1_b, nullptr, Q1, nullptr, M, D, D);
  k_gemm_bf16<<<gg, 256, 0, stream>>>(x1b, wb[1], k1_b, nullptr, K1, nullptr, M, D, D);
  k_gemm_bf16<<<gg, 256, 0, stream>>>(x1b, wb[2], v1_b, nullptr, nullptr, V1t, M, D, D);
  k_gemm_bf16<<<gg, 256, 0, stream>>>(x1b, wb[3], q2_b, nullptr, Q2, nullptr, M, D, D);
  k_gemm_bf16<<<gg, 256, 0, stream>>>(x2b, wb[4], k2_b, nullptr, K2, nullptr, M, D, D);
  k_gemm_bf16<<<gg, 256, 0, stream>>>(x2b, wb[5], v2_b, nullptr, nullptr, V2t, M, D, D);

  // 3) attentions
  dim3 ga(S / 32, B);
  k_attn<<<ga, 256, 0, stream>>>(Q1, K1, V1t, xw1, S, M);
  k_attn<<<ga, 256, 0, stream>>>(Q2, K2, V2t, xw2, S, M);

  // 4) h = LN(x1 + lam*xw1 + (1-lam)*xw2)
  k_ln<<<M / 8, 256, 0, stream>>>(x1, xw1, xw2, lam, 1, ln1g, ln1b, hf, hb, M);

  // 5) fc1: t = h @ fc1^T + b
  k_gemm_bf16<<<gg, 256, 0, stream>>>(hb, wb[6], fc1b, tb, nullptr, nullptr, M, D, D);

  // 6) h2 = LN(h + t)
  k_ln<<<M / 8, 256, 0, stream>>>(hf, tb, nullptr, nullptr, 0, ln2g, ln2b, nullptr, h2b, M);

  // 7) fc2 -> d_out (fp32)
  k_gemm_bf16<<<gg, 256, 0, stream>>>(h2b, wb[7], fc2b, (float*)d_out, nullptr, nullptr, M, D, D);
}
